// MakeTransformer_25984552141348
// MI455X (gfx1250) — matrix operand order repaired and checked by reading, not compiled, not run on hardware
//
#include <hip/hip_runtime.h>
#include <hip/hip_bf16.h>
#include <math.h>
#include <stdint.h>

// ---------------------------------------------------------------------------
// MI455X (gfx1250, wave32) transformer forward, round 4.
// All GEMMs on v_wmma_f32_16x16x32_f16, f32 accumulate.
//  - mt_gemm: 128x128 block tile, software-pipelined K loop with ping-pong
//    LDS buffers; A via GLOBAL_LOAD_ASYNC_TO_LDS (ASYNCcnt), B staged
//    transposed so fragment reads are ds_load_b128 pairs; one barrier/chunk;
//    global_prefetch two chunks ahead.
//  - Attention: padded dense per-(b,h) layouts, block per (b,h); Q/K/P staged
//    via async-to-LDS, V transposed through registers.
//  - Fused epilogues: bias / GELU / residual / f16 convert / head-scatter.
// ---------------------------------------------------------------------------

typedef __attribute__((ext_vector_type(16))) _Float16 v16h;
typedef __attribute__((ext_vector_type(8)))  float    v8f;
typedef _Float16 half_t;

#define BATCH 512
#define SEQ   71
#define SP    80            /* padded seq rows */
#define SPK   96            /* padded K-dim for p cols / v rows */
#define EMB   512
#define HEADS 8
#define DK    64
#define DFFN  2048
#define NLAY  4
#define OUTD  128
#define SFHID 128
#define MROWS (BATCH*SEQ)   /* 36352 = 284*128 */
#define BH    (BATCH*HEADS) /* 4096 */
#define THREADS 256
#define WPB     8

__device__ __forceinline__ float gelu_f(float x) {
    return 0.5f * x * (1.0f + erff(x * 0.70710678118654752f));
}

__device__ __forceinline__ v8f wmma_f16(v16h a, v16h b, v8f c) {
    return __builtin_amdgcn_wmma_f32_16x16x32_f16(
        false, a, false, b, (short)0, c, false, false);
}

// async global->LDS copy of 16 bytes per lane; tracked by ASYNCcnt
__device__ __forceinline__ void async_copy_b128(const half_t* g, half_t* lds) {
    unsigned la = (unsigned)(uintptr_t)lds;   // low 32 bits = LDS byte address
    asm volatile("global_load_async_to_lds_b128 %0, %1, off"
                 :: "v"(la), "v"(g) : "memory");
}
__device__ __forceinline__ void async_wait0() {
    asm volatile("s_wait_asynccnt 0x0" ::: "memory");
}

// ---- dense fragment gathers (layouts per CDNA5 ISA §7.12.2) ----
// A (16x32): lane -> M=lane%16, half=lane/16; contiguous 8+8 halfs per lane.
__device__ __forceinline__ v16h frag_a(const half_t* A, int lda, int m0, int k0) {
    int lane = threadIdx.x & 31;
    const half_t* p = A + (size_t)(m0 + (lane & 15)) * lda + k0 + ((lane >> 4) << 3);
    v16h a;
#pragma unroll
    for (int i = 0; i < 4; ++i) { a[2*i] = p[2*i]; a[2*i+1] = p[2*i+1]; }
    const half_t* p2 = p + 16;
#pragma unroll
    for (int i = 0; i < 4; ++i) { a[8+2*i] = p2[2*i]; a[9+2*i] = p2[2*i+1]; }
    return a;
}

// B from K-major storage Bt[n][k] (ldb = K stride): 8+8 halfs per lane (same k order as A).
__device__ __forceinline__ v16h frag_bt(const half_t* Bt, int ldb, int k0, int n0) {
    int lane = threadIdx.x & 31;
    const half_t* p = Bt + (size_t)(n0 + (lane & 15)) * ldb + k0 + ((lane >> 4) << 3);
    v16h b;
#pragma unroll
    for (int j = 0; j < 16; ++j) b[j] = p[j + ((j >> 3) << 3)];
    return b;
}

// ---------------------------------------------------------------------------
// Elementwise helpers
// ---------------------------------------------------------------------------
__global__ void mt_f32_to_f16(const float* __restrict__ s, half_t* __restrict__ d,
                              size_t n) {
    size_t i = (size_t)blockIdx.x * blockDim.x + threadIdx.x;
    size_t stride = (size_t)gridDim.x * blockDim.x;
    for (; i < n; i += stride) d[i] = (half_t)s[i];
}

__global__ void mt_zero_f16(half_t* __restrict__ d, size_t n) {
    size_t i = (size_t)blockIdx.x * blockDim.x + threadIdx.x;
    size_t stride = (size_t)gridDim.x * blockDim.x;
    for (; i < n; i += stride) d[i] = (half_t)0.f;
}

// zero pad rows [rowstart, rows) of BH matrices shaped [rows][DK]
__global__ void mt_zero_padrows(half_t* __restrict__ buf, int rows, int rowstart) {
    int span = (rows - rowstart) * DK;
    size_t n = (size_t)BH * span;
    size_t i = (size_t)blockIdx.x * blockDim.x + threadIdx.x;
    size_t stride = (size_t)gridDim.x * blockDim.x;
    for (; i < n; i += stride) {
        size_t bh = i / span;
        int off = (int)(i % span);
        buf[bh * ((size_t)rows * DK) + (size_t)rowstart * DK + off] = (half_t)0.f;
    }
}

// x + positional encoding -> f32 + f16 copies
__global__ void mt_addpos(const float* __restrict__ x, float* __restrict__ xf,
                          half_t* __restrict__ xh) {
    size_t i = (size_t)blockIdx.x * THREADS + threadIdx.x;  // exact grid
    int e = (int)(i % EMB);
    int s = (int)((i / EMB) % SEQ);
    float freq = __expf(-(float)(e & ~1) * (9.210340371976184f / (float)EMB));
    float ang = (float)s * freq;
    float pe = (e & 1) ? __cosf(ang) : __sinf(ang);
    float v = x[i] + pe;
    xf[i] = v;
    xh[i] = (half_t)v;
}

// ---------------------------------------------------------------------------
// Software-pipelined LDS-tiled WMMA GEMM: C = A*B + bias, fused epilogue.
// Block tile 128x128, K-chunk 32, ping-pong LDS buffers, one barrier/chunk.
// M%128==0, N%128==0, K%32==0.
// QKV=true: scatter-store f16 into padded per-head layout [BH][SP][DK].
// ---------------------------------------------------------------------------
template <bool GELU, bool RES, bool O32, bool O16, bool QKV>
__global__ void mt_gemm(const half_t* __restrict__ A, const half_t* __restrict__ B,
                        const float* __restrict__ bias, const float* __restrict__ resid,
                        float* __restrict__ C32, half_t* __restrict__ C16,
                        int Mn, int Nn, int Kn) {
    __shared__ __align__(16) half_t As[2 * 128 * 32];    // ping-pong [m][k]
    __shared__ __align__(16) half_t Bst[2 * 128 * 32];   // ping-pong [n][k]

    int tid = threadIdx.x;
    int wave = tid >> 5;
    int lane = tid & 31;
    int mblocks = Mn >> 7;
    int mt = blockIdx.x % mblocks;          // consecutive blocks share B (L2)
    int nblk = blockIdx.x / mblocks;

    v8f acc[8];
#pragma unroll
    for (int j = 0; j < 8; ++j) acc[j] = (v8f){0.f,0.f,0.f,0.f,0.f,0.f,0.f,0.f};

    auto stage_a = [&](int k0, half_t* dst) {
#pragma unroll
        for (int t = 0; t < 2; ++t) {
            int idx = tid + t * 256;
            int row = idx >> 2;             // 4 x b128 per 64B row
            int cu = idx & 3;
            async_copy_b128(A + (size_t)(mt * 128 + row) * Kn + k0 + cu * 8,
                            dst + row * 32 + cu * 8);
        }
    };
    auto stage_b = [&](int k0, half_t* dstB) {    // 2x2 transpose repack
        unsigned* du = (unsigned*)dstB;
#pragma unroll
        for (int t = 0; t < 4; ++t) {
            int blk = tid + t * 256;        // 0..1023 (2x2 half blocks)
            int n2 = blk & 63;
            int k2 = blk >> 6;
            const unsigned* g0 = (const unsigned*)(B + (size_t)(k0 + 2*k2) * Nn + nblk * 128);
            const unsigned* g1 = (const unsigned*)(B + (size_t)(k0 + 2*k2 + 1) * Nn + nblk * 128);
            unsigned u0 = g0[n2], u1 = g1[n2];
            du[(2*n2) * 16 + k2]   = (u0 & 0xffffu) | (u1 << 16);
            du[(2*n2+1) * 16 + k2] = (u0 >> 16) | (u1 & 0xffff0000u);
        }
    };

    int nchunks = Kn >> 5;
    stage_a(0, As);
    stage_b(0, Bst);
    async_wait0();
    __syncthreads();

    for (int ci = 0; ci < nchunks; ++ci) {
        int cur = ci & 1;
        half_t* Acur = As + cur * (128 * 32);
        half_t* Bcur = Bst + cur * (128 * 32);
        if (ci + 1 < nchunks) {
            int k0n = (ci + 1) << 5;
            stage_a(k0n, As + (cur ^ 1) * (128 * 32));
            stage_b(k0n, Bst + (cur ^ 1) * (128 * 32));
            if (ci + 2 < nchunks)
                __builtin_prefetch(B + (size_t)(((ci + 2) << 5) + (tid >> 6)) * Nn
                                     + nblk * 128 + (tid & 63) * 2, 0, 0);
        }
        v16h a = frag_a(Acur, 32, wave * 16, 0);
#pragma unroll
        for (int j = 0; j < 8; ++j) {
            v16h b = frag_bt(Bcur, 32, 0, j * 16);
            acc[j] = wmma_f16(a, b, acc[j]);
        }
        if (ci + 1 < nchunks) {
            async_wait0();
            __syncthreads();
        }
    }

    int half = lane >> 4;
#pragma unroll
    for (int j = 0; j < 8; ++j) {
        int n = nblk * 128 + j * 16 + (lane & 15);
        float bv = bias ? bias[n] : 0.f;
#pragma unroll
        for (int r = 0; r < 8; ++r) {
            int m = mt * 128 + wave * 16 + r + 8 * half;
            float v = acc[j][r] + bv;
            if (GELU) v = gelu_f(v);
            if (QKV) {
                int b = m / SEQ;
                int s = m - b * SEQ;
                int h = n >> 6;
                int d = n & (DK - 1);
                C16[(((size_t)b * HEADS + h) * SP + s) * DK + d] = (half_t)v;
            } else {
                size_t idx = (size_t)m * Nn + n;
                if (RES) v += resid[idx];
                if (O32) C32[idx] = v;
                if (O16) C16[idx] = (half_t)v;
            }
        }
    }
}

// ---------------------------------------------------------------------------
// scores = Q*K^T/8 per (b,h). Block per bh; Q,K async-staged to LDS (1280
// b128s, 5/thread, pointer-select keeps EXEC full); 8 waves sweep 25 tiles.
// ---------------------------------------------------------------------------
__global__ void mt_scores(const half_t* __restrict__ q, const half_t* __restrict__ k,
                          float* __restrict__ scores) {
    __shared__ __align__(16) half_t Qs[SP * DK];   // 10 KB
    __shared__ __align__(16) half_t Ks[SP * DK];
    int tid = threadIdx.x;
    int bh = blockIdx.x;
    const half_t* qb = q + (size_t)bh * SP * DK;
    const half_t* kb = k + (size_t)bh * SP * DK;
#pragma unroll
    for (int t = 0; t < 5; ++t) {
        int idx = tid + t * 256;            // 0..1279; 640 per matrix
        const half_t* src = (idx < 640) ? qb + idx * 8 : kb + (idx - 640) * 8;
        half_t*       dst = (idx < 640) ? Qs + idx * 8 : Ks + (idx - 640) * 8;
        async_copy_b128(src, dst);
    }
    async_wait0();
    __syncthreads();

    int wave = tid >> 5;
    int lane = tid & 31;
    int half = lane >> 4;
    float* out = scores + (size_t)bh * SP * SP;
    for (int t = wave; t < 25; t += WPB) {
        int mt = t / 5, nt = t % 5;
        v8f c = (v8f){0.f,0.f,0.f,0.f,0.f,0.f,0.f,0.f};
#pragma unroll
        for (int k0 = 0; k0 < DK; k0 += 32) {
            v16h a = frag_a(Qs, DK, mt * 16, k0);
            v16h b = frag_bt(Ks, DK, k0, nt * 16);
            c = wmma_f16(a, b, c);
        }
        int n = nt * 16 + (lane & 15);
#pragma unroll
        for (int r = 0; r < 8; ++r) {
            int m = mt * 16 + r + 8 * half;
            out[(size_t)m * SP + n] = c[r] * 0.125f;
        }
    }
}

// ---------------------------------------------------------------------------
// Fused: s' = s + gelu(s@W1+b1)@W2 + b2 ; mask ; softmax over 71 keys.
// One wave per score row; W1/W2 async-staged to LDS as f16. Exact grid.
// ---------------------------------------------------------------------------
__global__ void mt_softmax_resnet(const float* __restrict__ scores,
                                  const half_t* __restrict__ w1,
                                  const half_t* __restrict__ w2,
                                  const float* __restrict__ b1,
                                  const float* __restrict__ b2,
                                  const int* __restrict__ mask,
                                  half_t* __restrict__ p) {
    __shared__ __align__(16) half_t w1s[SEQ * SFHID];   // 9088 halfs = 1136 b128
    __shared__ __align__(16) half_t w2s[SFHID * SEQ];
    __shared__ float srow[WPB][SEQ + 1];
    __shared__ float hrow[WPB][SFHID];

    int tid = threadIdx.x;
#pragma unroll
    for (int t = 0; t < 9; ++t) {
        int idx = tid + t * 256;            // need 2272 of 2304
        if (idx < 2272) {
            const half_t* src = (idx < 1136) ? w1 + idx * 8 : w2 + (idx - 1136) * 8;
            half_t*       dst = (idx < 1136) ? w1s + idx * 8 : w2s + (idx - 1136) * 8;
            async_copy_b128(src, dst);
        }
    }
    async_wait0();
    __syncthreads();

    int wave = tid >> 5;
    int lane = tid & 31;
    int r = blockIdx.x * WPB + wave;        // < BH*SEQ (exact)
    int bh = r / SEQ;
    int sq = r % SEQ;
    int b = bh >> 3;
    const float* sp = scores + (size_t)bh * SP * SP + (size_t)sq * SP;

    float sreg[3];
#pragma unroll
    for (int t = 0; t < 3; ++t) {
        int j = lane + 32 * t;
        float v = (j < SEQ) ? sp[j] : 0.f;
        sreg[t] = v;
        if (j < SEQ) srow[wave][j] = v;
    }
    __syncthreads();

#pragma unroll
    for (int t = 0; t < 4; ++t) {
        int n = lane + 32 * t;
        float acc = b1[n];
        for (int j = 0; j < SEQ; ++j)
            acc += srow[wave][j] * (float)w1s[j * SFHID + n];
        hrow[wave][n] = gelu_f(acc);
    }
    __syncthreads();

    float v[3];
#pragma unroll
    for (int t = 0; t < 3; ++t) {
        int j = lane + 32 * t;
        if (j < SEQ) {
            float o = b2[j];
            for (int n = 0; n < SFHID; ++n)
                o += hrow[wave][n] * (float)w2s[n * SEQ + j];
            float sc = sreg[t] + o;
            if (mask[(size_t)b * SEQ + j] == 0) sc = -1e9f;
            v[t] = sc;
        } else {
            v[t] = -INFINITY;
        }
    }
    float mx = fmaxf(fmaxf(v[0], v[1]), v[2]);
#pragma unroll
    for (int d = 16; d; d >>= 1) mx = fmaxf(mx, __shfl_xor(mx, d, 32));
    float es[3];
    float sum = 0.f;
#pragma unroll
    for (int t = 0; t < 3; ++t) {
        int j = lane + 32 * t;
        es[t] = (j < SEQ) ? __expf(v[t] - mx) : 0.f;
        sum += es[t];
    }
#pragma unroll
    for (int d = 16; d; d >>= 1) sum += __shfl_xor(sum, d, 32);
    float inv = 1.f / sum;
    half_t* pp = p + (size_t)bh * SP * SPK + (size_t)sq * SPK;
#pragma unroll
    for (int t = 0; t < 3; ++t) {
        int j = lane + 32 * t;
        if (j < SEQ) pp[j] = (half_t)(es[t] * inv);
    }
}

// ---------------------------------------------------------------------------
// attn = P @ V per (b,h). Block per bh; P async-staged (960 b128s), V staged
// transposed as Vst[n][k] so fragments are ds_load_b128 pairs.
// Store-side guard m<SEQ only; scatter back to [B,S,E].
// ---------------------------------------------------------------------------
__global__ void mt_attn_v(const half_t* __restrict__ p, const half_t* __restrict__ v,
                          half_t* __restrict__ attn) {
    __shared__ __align__(16) half_t Ps[SP * SPK];    // 15 KB
    __shared__ __align__(16) half_t Vst[DK * SPK];   // 12 KB, [n][k]
    int tid = threadIdx.x;
    int bh = blockIdx.x;
    int h = bh & (HEADS - 1);
    int b = bh >> 3;
    const half_t* pb = p + (size_t)bh * SP * SPK;
#pragma unroll
    for (int t = 0; t < 4; ++t) {
        int idx = tid + t * 256;            // need 960 of 1024
        if (idx < 960) async_copy_b128(pb + idx * 8, Ps + idx * 8);
    }
    const half_t* vb = v + (size_t)bh * SPK * DK;
    unsigned* Vstu = (unsigned*)Vst;
#pragma unroll
    for (int t = 0; t < 6; ++t) {           // 1536 2x2 half blocks
        int blk = tid + t * 256;
        int n2 = blk & 31;                  // 32 col-pairs (DK/2)
        int k2 = blk >> 5;                  // 48 row-pairs (SPK/2)
        const unsigned* g0 = (const unsigned*)(vb + (size_t)(2*k2) * DK);
        const unsigned* g1 = (const unsigned*)(vb + (size_t)(2*k2 + 1) * DK);
        unsigned u0 = g0[n2], u1 = g1[n2];
        Vstu[(2*n2) * 48 + k2]   = (u0 & 0xffffu) | (u1 << 16);
        Vstu[(2*n2+1) * 48 + k2] = (u0 >> 16) | (u1 & 0xffff0000u);
    }
    async_wait0();
    __syncthreads();

    int wave = tid >> 5;
    int lane = tid & 31;
    int half = lane >> 4;
    for (int t = wave; t < 20; t += WPB) {
        int mt = t >> 2, nt = t & 3;
        v8f c = (v8f){0.f,0.f,0.f,0.f,0.f,0.f,0.f,0.f};
#pragma unroll
        for (int k0 = 0; k0 < SPK; k0 += 32) {
            v16h a = frag_a(Ps, SPK, mt * 16, k0);
            v16h bb = frag_bt(Vst, SPK, k0, nt * 16);
            c = wmma_f16(a, bb, c);
        }
        int n = nt * 16 + (lane & 15);
#pragma unroll
        for (int r = 0; r < 8; ++r) {
            int m = mt * 16 + r + 8 * half;
            if (m < SEQ)
                attn[((size_t)b * SEQ + m) * EMB + h * DK + n] = (half_t)c[r];
        }
    }
}

// ---------------------------------------------------------------------------
// Final LayerNorm (ddof=1, eps on std) + GELU -> f16. One wave per row.
// ---------------------------------------------------------------------------
__global__ void mt_ln_gelu(const float* __restrict__ x, const float* __restrict__ ga,
                           const float* __restrict__ gb, half_t* __restrict__ y) {
    int wave = threadIdx.x >> 5;
    int lane = threadIdx.x & 31;
    size_t row = (size_t)blockIdx.x * WPB + wave;   // exact grid
    const float* xp = x + row * EMB;
    float vals[16];
    float s = 0.f;
#pragma unroll
    for (int i = 0; i < 16; ++i) {
        vals[i] = xp[i * 32 + lane];
        s += vals[i];
    }
#pragma unroll
    for (int d = 16; d; d >>= 1) s += __shfl_xor(s, d, 32);
    float mean = s * (1.f / EMB);
    float q = 0.f;
#pragma unroll
    for (int i = 0; i < 16; ++i) {
        float dlt = vals[i] - mean;
        q += dlt * dlt;
    }
#pragma unroll
    for (int d = 16; d; d >>= 1) q += __shfl_xor(q, d, 32);
    float stdv = sqrtf(q * (1.f / (EMB - 1)));
    float inv = 1.f / (stdv + 1e-6f);
#pragma unroll
    for (int i = 0; i < 16; ++i) {
        int e = i * 32 + lane;
        float t = ga[e] * (vals[i] - mean) * inv + gb[e];
        y[row * EMB + e] = (half_t)gelu_f(t);
    }
}

// ---------------------------------------------------------------------------
// Workspace layout. FFN hidden (149 MB) aliases q/k/v/scores (first 239 MB of
// the attention region); p (zeroed once) lies beyond the alias span. Pad rows
// of q/k/v are re-zeroed each layer.
// ---------------------------------------------------------------------------
static constexpr size_t AL(size_t x) { return (x + 255) & ~(size_t)255; }
static constexpr size_t SZ_W    = (size_t)NLAY * EMB * EMB * 2;
static constexpr size_t SZ_FFW  = (size_t)NLAY * EMB * DFFN * 2;
static constexpr size_t SZ_SFW  = (size_t)NLAY * SEQ * SFHID * 2;
static constexpr size_t SZ_OUTW = (size_t)EMB * OUTD * 2;
static constexpr size_t SZ_XF   = (size_t)MROWS * EMB * 4;
static constexpr size_t SZ_XH   = (size_t)MROWS * EMB * 2;
static constexpr size_t SZ_QK   = (size_t)BH * SP * DK * 2;
static constexpr size_t SZ_V    = (size_t)BH * SPK * DK * 2;
static constexpr size_t SZ_SC   = (size_t)BH * SP * SP * 4;
static constexpr size_t SZ_P    = (size_t)BH * SP * SPK * 2;

static constexpr size_t O_WQ   = 0;
static constexpr size_t O_WK   = AL(O_WQ + SZ_W);
static constexpr size_t O_WV   = AL(O_WK + SZ_W);
static constexpr size_t O_WO   = AL(O_WV + SZ_W);
static constexpr size_t O_FF1  = AL(O_WO + SZ_W);
static constexpr size_t O_FF2  = AL(O_FF1 + SZ_FFW);
static constexpr size_t O_SF1  = AL(O_FF2 + SZ_FFW);
static constexpr size_t O_SF2  = AL(O_SF1 + SZ_SFW);
static constexpr size_t O_OUTW = AL(O_SF2 + SZ_SFW);
static constexpr size_t O_XF   = AL(O_OUTW + SZ_OUTW);
static constexpr size_t O_XH   = AL(O_XF + SZ_XF);
static constexpr size_t O_Q    = AL(O_XH + SZ_XH);
static constexpr size_t O_K    = AL(O_Q + SZ_QK);
static constexpr size_t O_V    = AL(O_K + SZ_QK);
static constexpr size_t O_SC   = AL(O_V + SZ_V);
static constexpr size_t O_P    = AL(O_SC + SZ_SC);
static constexpr size_t O_ATT  = AL(O_P + SZ_P);
static constexpr size_t O_H    = O_Q;   // alias: q+k+v+scores span > 149 MB
static constexpr size_t O_Y    = O_Q;   // alias: final phase
static constexpr size_t WS_NEED = AL(O_ATT + SZ_XH);

extern "C" void kernel_launch(void* const* d_in, const int* in_sizes, int n_in,
                              void* d_out, int out_size, void* d_ws, size_t ws_size,
                              hipStream_t stream) {
    (void)in_sizes; (void)n_in; (void)out_size;
    if (ws_size < WS_NEED) return;

    const float* x      = (const float*)d_in[0];
    const int*   mask   = (const int*)d_in[1];
    const float* wq     = (const float*)d_in[3];
    const float* bq     = (const float*)d_in[4];
    const float* wk     = (const float*)d_in[5];
    const float* bk     = (const float*)d_in[6];
    const float* wv     = (const float*)d_in[7];
    const float* bv     = (const float*)d_in[8];
    const float* wo     = (const float*)d_in[9];
    const float* bo     = (const float*)d_in[10];
    const float* sfw1   = (const float*)d_in[11];
    const float* sfb1   = (const float*)d_in[12];
    const float* sfw2   = (const float*)d_in[13];
    const float* sfb2   = (const float*)d_in[14];
    const float* ffw1   = (const float*)d_in[15];
    const float* ffb1   = (const float*)d_in[16];
    const float* ffw2   = (const float*)d_in[17];
    const float* ffb2   = (const float*)d_in[18];
    const float* norm_a = (const float*)d_in[19];
    const float* norm_b = (const float*)d_in[20];
    const float* out_w  = (const float*)d_in[21];
    const float* out_b  = (const float*)d_in[22];
    float* outp = (float*)d_out;

    char* ws = (char*)d_ws;
    half_t* wqh  = (half_t*)(ws + O_WQ);
    half_t* wkh  = (half_t*)(ws + O_WK);
    half_t* wvh  = (half_t*)(ws + O_WV);
    half_t* woh  = (half_t*)(ws + O_WO);
    half_t* ff1h = (half_t*)(ws + O_FF1);
    half_t* ff2h = (half_t*)(ws + O_FF2);
    half_t* sf1h = (half_t*)(ws + O_SF1);
    half_t* sf2h = (half_t*)(ws + O_SF2);
    half_t* outwh= (half_t*)(ws + O_OUTW);
    float*  xf   = (float*)(ws + O_XF);
    half_t* xh   = (half_t*)(ws + O_XH);
    half_t* qh   = (half_t*)(ws + O_Q);
    half_t* kh   = (half_t*)(ws + O_K);
    half_t* vh   = (half_t*)(ws + O_V);
    float*  sc   = (float*)(ws + O_SC);
    half_t* ph   = (half_t*)(ws + O_P);
    half_t* atth = (half_t*)(ws + O_ATT);
    half_t* hh   = (half_t*)(ws + O_H);
    half_t* yh   = (half_t*)(ws + O_Y);

    auto cvt = [&](const float* s, half_t* d, size_t n) {
        int blocks = (int)((n + THREADS - 1) / THREADS);
        if (blocks > 65536) blocks = 65536;
        mt_f32_to_f16<<<blocks, THREADS, 0, stream>>>(s, d, n);
    };
    cvt(wq,   wqh,  (size_t)NLAY * EMB * EMB);
    cvt(wk,   wkh,  (size_t)NLAY * EMB * EMB);
    cvt(wv,   wvh,  (size_t)NLAY * EMB * EMB);
    cvt(wo,   woh,  (size_t)NLAY * EMB * EMB);
    cvt(ffw1, ff1h, (size_t)NLAY * EMB * DFFN);
    cvt(ffw2, ff2h, (size_t)NLAY * DFFN * EMB);
    cvt(sfw1, sf1h, (size_t)NLAY * SEQ * SFHID);
    cvt(sfw2, sf2h, (size_t)NLAY * SFHID * SEQ);
    cvt(out_w, outwh, (size_t)EMB * OUTD);

    // p padded buffer: zero once (pad cols/rows never touched afterwards)
    mt_zero_f16<<<65536, THREADS, 0, stream>>>(ph, (size_t)BH * SP * SPK);

    // x + positional encoding
    mt_addpos<<<(MROWS * EMB) / THREADS, THREADS, 0, stream>>>(x, xf, xh);

    auto gemm_grid = [](int M, int N) { return (M >> 7) * (N >> 7); };

    for (int i = 0; i < NLAY; ++i) {
        const half_t* Wq = wqh + (size_t)i * EMB * EMB;
        const half_t* Wk = wkh + (size_t)i * EMB * EMB;
        const half_t* Wv = wvh + (size_t)i * EMB * EMB;
        const half_t* Wo = woh + (size_t)i * EMB * EMB;
        const half_t* F1 = ff1h + (size_t)i * EMB * DFFN;
        const half_t* F2 = ff2h + (size_t)i * DFFN * EMB;
        const half_t* S1 = sf1h + (size_t)i * SEQ * SFHID;
        const half_t* S2 = sf2h + (size_t)i * SFHID * SEQ;

        // re-zero pad rows (clobbered by FFN-hidden alias last layer)
        mt_zero_padrows<<<2048, THREADS, 0, stream>>>(qh, SP, SEQ);
        mt_zero_padrows<<<2048, THREADS, 0, stream>>>(kh, SP, SEQ);
        mt_zero_padrows<<<2048, THREADS, 0, stream>>>(vh, SPK, SEQ);

        // Q, K, V projections, scatter into padded [BH][SP(K)][DK] f16
        mt_gemm<false, false, false, true, true><<<gemm_grid(MROWS, EMB), THREADS, 0, stream>>>(
            xh, Wq, bq + (size_t)i * EMB, nullptr, nullptr, qh, MROWS, EMB, EMB);
        mt_gemm<false, false, false, true, true><<<gemm_grid(MROWS, EMB), THREADS, 0, stream>>>(
            xh, Wk, bk + (size_t)i * EMB, nullptr, nullptr, kh, MROWS, EMB, EMB);
        mt_gemm<false, false, false, true, true><<<gemm_grid(MROWS, EMB), THREADS, 0, stream>>>(
            xh, Wv, bv + (size_t)i * EMB, nullptr, nullptr, vh, MROWS, EMB, EMB);

        // scores = QK^T / sqrt(dk); fused score-MLP + mask + softmax; P@V
        mt_scores<<<BH, THREADS, 0, stream>>>(qh, kh, sc);
        mt_softmax_resnet<<<(BH * SEQ) / WPB, THREADS, 0, stream>>>(
            sc, S1, S2, sfb1 + (size_t)i * SFHID, sfb2 + (size_t)i * SEQ, mask, ph);
        mt_attn_v<<<BH, THREADS, 0, stream>>>(ph, vh, atth);

        // x = x + attn @ Wo + bo
        mt_gemm<false, true, true, true, false><<<gemm_grid(MROWS, EMB), THREADS, 0, stream>>>(
            atth, Wo, bo + (size_t)i * EMB, xf, xf, xh, MROWS, EMB, EMB);

        // h = gelu(x @ ffw1 + ffb1)   (aliases q/k/v/scores scratch)
        mt_gemm<true, false, false, true, false><<<gemm_grid(MROWS, DFFN), THREADS, 0, stream>>>(
            xh, F1, ffb1 + (size_t)i * DFFN, nullptr, nullptr, hh, MROWS, DFFN, EMB);

        // x = x + h @ ffw2 + ffb2
        mt_gemm<false, true, true, true, false><<<gemm_grid(MROWS, EMB), THREADS, 0, stream>>>(
            hh, F2, ffb2 + (size_t)i * EMB, xf, xf, xh, MROWS, EMB, DFFN);
    }

    // final layernorm + gelu -> y (f16), then out GEMM -> d_out (f32)
    mt_ln_gelu<<<MROWS / WPB, THREADS, 0, stream>>>(xf, norm_a, norm_b, yh);
    mt_gemm<false, false, true, false, false><<<gemm_grid(MROWS, OUTD), THREADS, 0, stream>>>(
        yh, outwh, out_b, nullptr, outp, nullptr, MROWS, OUTD, EMB);
}
